// QuantizedLinear_6459630813322
// MI455X (gfx1250) — 8-bit matrix operand order repaired and checked by reading, compiled, not run on hardware
//
#include <hip/hip_runtime.h>
#include <cstdint>
#include <cfloat>

typedef __attribute__((ext_vector_type(8))) int v8i;

#define D_K 4096
#define D_N 4096

__global__ void qlin_init(float* ws) {
    ws[0] = FLT_MAX;  ws[1] = -FLT_MAX;
    ws[2] = FLT_MAX;  ws[3] = -FLT_MAX;
}

__global__ void qlin_minmax(const float4* __restrict__ src, int n4,
                            float* __restrict__ mn_out, float* __restrict__ mx_out) {
    float mn = FLT_MAX, mx = -FLT_MAX;
    const int stride = gridDim.x * blockDim.x;
    for (int i = blockIdx.x * blockDim.x + threadIdx.x; i < n4; i += stride) {
        const float4 v = src[i];
        mn = fminf(mn, fminf(fminf(v.x, v.y), fminf(v.z, v.w)));
        mx = fmaxf(mx, fmaxf(fmaxf(v.x, v.y), fmaxf(v.z, v.w)));
    }
    __shared__ float smn[256];
    __shared__ float smx[256];
    const int t = threadIdx.x;
    smn[t] = mn; smx[t] = mx;
    __syncthreads();
    for (int s = 128; s > 0; s >>= 1) {
        if (t < s) {
            smn[t] = fminf(smn[t], smn[t + s]);
            smx[t] = fmaxf(smx[t], smx[t + s]);
        }
        __syncthreads();
    }
    if (t == 0) {
        atomicMin(mn_out, smn[0]);
        atomicMax(mx_out, smx[0]);
    }
}

__global__ void qlin_params(float* ws) {
    const float xmin = ws[0], xmax = ws[1], wmin = ws[2], wmax = ws[3];
    const float sx = (xmax - xmin) * (1.0f / 255.0f);
    const float zx = rintf(-128.0f - xmin / sx);
    const float sw = (wmax - wmin) * (1.0f / 255.0f);
    const float zw = rintf(-128.0f - wmin / sw);
    ws[4] = sx; ws[5] = zx; ws[6] = sw; ws[7] = zw; ws[8] = sx * sw;
}

__global__ void qlin_quant_rowsum(const float* __restrict__ src,
                                  int8_t* __restrict__ dst,
                                  int* __restrict__ rowsum,
                                  const float* __restrict__ params, int pidx) {
    const int row = blockIdx.x;
    const int t = threadIdx.x;
    const float inv_s = 1.0f / params[pidx];
    const float zp = params[pidx + 1];
    const int base = row * D_K;
    int rsum = 0;
#pragma unroll
    for (int j = 0; j < D_K / 256; ++j) {
        const int idx = base + j * 256 + t;
        float q = rintf(src[idx] * inv_s) + zp;
        q = fminf(fmaxf(q, -128.0f), 127.0f);
        const int qi = (int)q;
        dst[idx] = (int8_t)qi;
        rsum += qi;
    }
    __shared__ int ssum[256];
    ssum[t] = rsum;
    __syncthreads();
    for (int s = 128; s > 0; s >>= 1) {
        if (t < s) ssum[t] += ssum[t + s];
        __syncthreads();
    }
    if (t == 0) rowsum[row] = ssum[0];
}

__global__ __launch_bounds__(256)
void qlin_gemm_wmma_i8(const int8_t* __restrict__ qx, const int8_t* __restrict__ qw,
                       const int* __restrict__ rsx, const int* __restrict__ rsw,
                       const float* __restrict__ params, const float* __restrict__ bias,
                       float* __restrict__ out) {
    __shared__ __align__(16) int8_t lds[2 * 16384];

    const int bm = blockIdx.y * 128;
    const int bn = blockIdx.x * 128;
    const int t = threadIdx.x;
    const int wave = t >> 5;
    const int lane = t & 31;
    const int wm = (wave >> 2) * 64;
    const int wn = (wave & 3) * 32;
    const int half = lane >> 4;
    const int r = lane & 15;

    v8i acc[4][2] = {};

    const int crow = t >> 1;
    const int coff = (t & 1) * 32;
    const unsigned lds_base = (unsigned)(uintptr_t)(void*)lds;
    const unsigned a_lds = lds_base + (unsigned)(crow * 64 + coff);
    const unsigned b_lds = a_lds + 8192u;
    const unsigned ga_off = (unsigned)((bm + crow) * D_K + coff);
    const unsigned gb_off = (unsigned)((bn + crow) * D_K + coff);

    auto stage = [&](int buf, int k0) {
        const unsigned al = a_lds + (unsigned)(buf * 16384);
        const unsigned bl = b_lds + (unsigned)(buf * 16384);
        const unsigned ao = ga_off + (unsigned)k0;
        const unsigned bo = gb_off + (unsigned)k0;
        asm volatile("global_load_async_to_lds_b128 %0, %1, %2"
                     :: "v"(al),        "v"(ao),        "s"(qx) : "memory");
        asm volatile("global_load_async_to_lds_b128 %0, %1, %2"
                     :: "v"(al + 16u),  "v"(ao + 16u),  "s"(qx) : "memory");
        asm volatile("global_load_async_to_lds_b128 %0, %1, %2"
                     :: "v"(bl),        "v"(bo),        "s"(qw) : "memory");
        asm volatile("global_load_async_to_lds_b128 %0, %1, %2"
                     :: "v"(bl + 16u),  "v"(bo + 16u),  "s"(qw) : "memory");
    };

    stage(0, 0);

    int buf = 0;
    for (int k0 = 0; k0 < D_K; k0 += 64, buf ^= 1) {
        stage(buf ^ 1, (k0 + 64) & (D_K - 1));
        asm volatile("s_wait_asynccnt 0x4" ::: "memory");
        __syncthreads();

        const int8_t* As = lds + buf * 16384;
        const int8_t* Bs = As + 8192;

        v8i bfrag[2];
#pragma unroll
        for (int nt = 0; nt < 2; ++nt) {
            const int8_t* bp = Bs + (wn + nt * 16 + r) * 64 + half * 16;
            const uint4 lo = *(const uint4*)bp;
            const uint4 hi = *(const uint4*)(bp + 32);
            v8i bf;
            bf[0] = (int)lo.x; bf[1] = (int)lo.y; bf[2] = (int)lo.z; bf[3] = (int)lo.w;
            bf[4] = (int)hi.x; bf[5] = (int)hi.y; bf[6] = (int)hi.z; bf[7] = (int)hi.w;
            bfrag[nt] = bf;
        }
#pragma unroll
        for (int mt = 0; mt < 4; ++mt) {
            const int8_t* ap = As + (wm + mt * 16 + r) * 64 + half * 16;
            const uint2 p0 = *(const uint2*)(ap);
            const uint2 p1 = *(const uint2*)(ap + 8);
            const uint2 p2 = *(const uint2*)(ap + 32);
            const uint2 p3 = *(const uint2*)(ap + 40);
            v8i af;
            af[0] = (int)p0.x; af[1] = (int)p0.y; af[2] = (int)p1.x; af[3] = (int)p1.y;
            af[4] = (int)p2.x; af[5] = (int)p2.y; af[6] = (int)p3.x; af[7] = (int)p3.y;
            acc[mt][0] = __builtin_amdgcn_wmma_i32_16x16x64_iu8(
                true, af, true, bfrag[0], acc[mt][0], false, false);
            acc[mt][1] = __builtin_amdgcn_wmma_i32_16x16x64_iu8(
                true, af, true, bfrag[1], acc[mt][1], false, false);
        }
    }

    const float sxsw = params[8];
    const int zxi = (int)params[5];
    const int zwi = (int)params[7];
    const int kzz = D_K * zxi * zwi;
#pragma unroll
    for (int mt = 0; mt < 4; ++mt) {
#pragma unroll
        for (int nt = 0; nt < 2; ++nt) {
            const int n = bn + wn + nt * 16 + r;
            const float bv = bias[n];
            const int rswn = rsw[n];
#pragma unroll
            for (int i = 0; i < 8; ++i) {
                const int m = bm + wm + mt * 16 + half * 8 + i;
                const int corr = acc[mt][nt][i] - zwi * rsx[m] - zxi * rswn + kzz;
                out[(size_t)m * D_N + n] = sxsw * (float)corr + bv;
            }
        }
    }
}

extern "C" void kernel_launch(void* const* d_in, const int* in_sizes, int n_in,
                              void* d_out, int out_size, void* d_ws, size_t ws_size,
                              hipStream_t stream) {
    const float* x    = (const float*)d_in[0];
    const float* W    = (const float*)d_in[1];
    const float* bias = (const float*)d_in[2];
    float* out = (float*)d_out;

    const int M = in_sizes[0] / D_K;
    const int N = D_N;

    char* ws = (char*)d_ws;
    float*  scal = (float*)ws;
    int8_t* qx   = (int8_t*)(ws + 256);
    int8_t* qw   = qx + (size_t)M * D_K;
    int*    rsx  = (int*)(qw + (size_t)N * D_K);
    int*    rsw  = rsx + M;

    qlin_init<<<1, 1, 0, stream>>>(scal);
    qlin_minmax<<<1024, 256, 0, stream>>>((const float4*)x, in_sizes[0] / 4,
                                          scal + 0, scal + 1);
    qlin_minmax<<<512, 256, 0, stream>>>((const float4*)W, in_sizes[1] / 4,
                                         scal + 2, scal + 3);
    qlin_params<<<1, 1, 0, stream>>>(scal);
    qlin_quant_rowsum<<<M, 256, 0, stream>>>(x, qx, rsx, scal, 4);
    qlin_quant_rowsum<<<N, 256, 0, stream>>>(W, qw, rsw, scal, 6);

    dim3 grid(N / 128, M / 128);
    qlin_gemm_wmma_i8<<<grid, 256, 0, stream>>>(qx, qw, rsx, rsw, scal, bias, out);
}
